// Rwkv_Tmix_x060_64347200029091
// MI455X (gfx1250) — matrix operand order repaired and checked by reading, not compiled, not run on hardware
//
#include <hip/hip_runtime.h>
#include <hip/hip_bf16.h>
#include <math.h>

// Problem constants (RWKV x060 time-mix)
#define BB 4
#define TT 4096
#define CC 2048
#define HH 32
#define NHEAD 64
#define MM (BB*TT)          // 16384 rows
#define DMIXC 32
#define DDECC 64
#define GN_EPS 6.4e-4f      // 1e-5 * 8^2

typedef __bf16 bhalf;
typedef __attribute__((ext_vector_type(16))) __bf16 v16bf;
typedef __attribute__((ext_vector_type(8)))  __bf16 v8bf;
typedef __attribute__((ext_vector_type(8)))  float  v8f;

__device__ __forceinline__ bhalf f2b(float f){ return (bhalf)f; }
__device__ __forceinline__ float b2f(bhalf h){ return (float)h; }
__device__ __forceinline__ v16bf cat8(v8bf lo, v8bf hi){
  return __builtin_shufflevector(lo, hi, 0,1,2,3,4,5,6,7,8,9,10,11,12,13,14,15);
}

// ---------------------------------------------------------------------------
// Weight convert + transpose: Bt[n*K+k] = (bf16)W[k*N+n]
// ---------------------------------------------------------------------------
__global__ void k_transpose_bf16(const float* __restrict__ W, bhalf* __restrict__ Bt,
                                 int K, int N){
  __shared__ float tile[32][33];
  int nb = blockIdx.x*32, kb = blockIdx.y*32;
  int tx = threadIdx.x & 31, ty = threadIdx.x >> 5;   // 256 thr: ty 0..7
  #pragma unroll
  for (int i = ty; i < 32; i += 8) tile[i][tx] = W[(size_t)(kb+i)*N + nb+tx];
  __syncthreads();
  #pragma unroll
  for (int i = ty; i < 32; i += 8) Bt[(size_t)(nb+i)*K + kb+tx] = f2b(tile[tx][i]);
}

// ---------------------------------------------------------------------------
// Token shift: xxx = x + (x_prev - x)*maa_x   (bf16 out, A-matrix for t5 GEMM)
// ---------------------------------------------------------------------------
__global__ void k_shift_mix(const float* __restrict__ x, const float* __restrict__ shift,
                            const float* __restrict__ maa_x, bhalf* __restrict__ xxx){
  size_t i = (size_t)blockIdx.x*256 + threadIdx.x;    // < MM*CC
  int c = (int)(i & (CC-1));
  size_t row = i >> 11;                               // / CC
  int t = (int)(row & (TT-1));
  int b = (int)(row >> 12);                           // / TT
  float xv = x[i];
  float prev = t ? x[i - CC] : shift[(size_t)b*CC + c];
  xxx[i] = f2b(xv + (prev - xv)*maa_x[c]);
}

// ---------------------------------------------------------------------------
// WMMA GEMM:  out[M,N] = A[M,K](bf16) * Bt[N,K]^T(bf16), f32 accumulate.
// Block = 256 threads = 8 waves; wave owns a (16*MT) x (16*NT) register tile:
// rows [by*128*MT + wave*16*MT, ...), cols [bx*16*NT, ...).
// Per K=32 step: MT A-fragments + NT B-fragments feed MT*NT WMMAs
// (B fragments reused across MT row-tiles -> fewer loads per WMMA).
// Fragment layouts follow CDNA5 ISA 7.12.2:
//   A 16x32 bf16: lane L<16 -> row L, K[kb..kb+7] and K[kb+16..kb+23]
//                 lane L>=16 -> row L-16, K[kb+8..]   and K[kb+24..]
//   B 32x16 bf16: lane = column; lane<16 -> K[kb..kb+7] and K[kb+16..kb+23], lane>=16 -> K[kb+8..] and K[kb+24..]
//   C/D: VGPR j -> row j + 8*(lane>=16), col = lane&15
// EPI: 0 f32 store, 1 tanh->bf16, 2 silu->bf16, 3 exp(-exp(vecN+v))->f32, 4 bf16
// ---------------------------------------------------------------------------
template<int MT, int NT, int EPI>
__global__ void k_gemm(const bhalf* __restrict__ A, const bhalf* __restrict__ Bt,
                       int M, int N, int K,
                       float* __restrict__ outF, bhalf* __restrict__ outB,
                       const float* __restrict__ vecN){
  int lane = threadIdx.x & 31;
  int wave = threadIdx.x >> 5;
  int row0 = blockIdx.y*(128*MT) + wave*(16*MT);
  int col0 = blockIdx.x*(16*NT);
  int lo = lane & 15, hi = lane >> 4;

  const bhalf* ap = A + (size_t)(row0+lo)*K + hi*8;
  v8f acc[MT][NT];
  #pragma unroll
  for (int mi = 0; mi < MT; mi++)
    #pragma unroll
    for (int t = 0; t < NT; t++) acc[mi][t] = (v8f){0,0,0,0,0,0,0,0};

  for (int kb = 0; kb < K; kb += 32){
    v16bf af[MT];
    #pragma unroll
    for (int mi = 0; mi < MT; mi++){
      const bhalf* am = ap + (size_t)(mi*16)*K + kb;
      af[mi] = cat8(*(const v8bf*)am, *(const v8bf*)(am + 16));
    }
    #pragma unroll
    for (int t = 0; t < NT; t++){
      const bhalf* bp = Bt + (size_t)(col0 + t*16 + lo)*K + kb + hi*8;
      v16bf bfr = cat8(*(const v8bf*)bp, *(const v8bf*)(bp + 16));
      #pragma unroll
      for (int mi = 0; mi < MT; mi++)
        acc[mi][t] = __builtin_amdgcn_wmma_f32_16x16x32_bf16(
            false, af[mi], false, bfr, (short)0, acc[mi][t], false, false);
    }
  }

  #pragma unroll
  for (int mi = 0; mi < MT; mi++){
    #pragma unroll
    for (int t = 0; t < NT; t++){
      int cc = col0 + t*16 + lo;
      #pragma unroll
      for (int j = 0; j < 8; j++){
        int rr = row0 + mi*16 + hi*8 + j;
        size_t idx = (size_t)rr*N + cc;
        float vv = acc[mi][t][j];
        if (EPI == 0)      outF[idx] = vv;
        else if (EPI == 1) outB[idx] = f2b(tanhf(vv));
        else if (EPI == 2) outB[idx] = f2b(vv / (1.f + __expf(-vv)));
        else if (EPI == 3) outF[idx] = __expf(-__expf(vecN[cc] + vv));
        else               outB[idx] = f2b(vv);
      }
    }
  }
}

// ---------------------------------------------------------------------------
// Combine the 5 data-dependent mixes (K=32 dot per lane, w2 column-coalesced)
// ---------------------------------------------------------------------------
__global__ void k_mix_combine(const float* __restrict__ x, const float* __restrict__ shift,
    const bhalf* __restrict__ t5, const float* __restrict__ w2,
    const float* __restrict__ mw, const float* __restrict__ mk, const float* __restrict__ mv,
    const float* __restrict__ mr, const float* __restrict__ mg,
    bhalf* __restrict__ xw, bhalf* __restrict__ xk, bhalf* __restrict__ xv,
    bhalf* __restrict__ xr, bhalf* __restrict__ xg){
  int bi = blockIdx.x;            // MM * (CC/256) blocks
  size_t m = (size_t)(bi >> 3);
  int c = ((bi & 7) << 8) + threadIdx.x;
  __shared__ float t5s[5*DMIXC];
  if (threadIdx.x < 5*DMIXC) t5s[threadIdx.x] = b2f(t5[m*(5*DMIXC) + threadIdx.x]);
  __syncthreads();
  int t = (int)(m & (TT-1));
  int b = (int)(m >> 12);
  size_t i = m*CC + c;
  float xval = x[i];
  float prev = t ? x[i - CC] : shift[(size_t)b*CC + c];
  float xx = prev - xval;
  float s[5];
  #pragma unroll
  for (int f = 0; f < 5; f++){
    const float* w2f = w2 + ((size_t)f*DMIXC)*CC + c;
    float a = 0.f;
    #pragma unroll
    for (int d = 0; d < DMIXC; d++) a = fmaf(t5s[f*DMIXC + d], w2f[(size_t)d*CC], a);
    s[f] = a;
  }
  xw[i] = f2b(xval + xx*(mw[c] + s[0]));
  xk[i] = f2b(xval + xx*(mk[c] + s[1]));
  xv[i] = f2b(xval + xx*(mv[c] + s[2]));
  xr[i] = f2b(xval + xx*(mr[c] + s[3]));
  xg[i] = f2b(xval + xx*(mg[c] + s[4]));
}

// ---------------------------------------------------------------------------
// Sequential WKV scan. One 64-thread block per (b,h); thread owns state
// column v: S[k][v] in 64 VGPRs. r/k/decay broadcast per step via LDS.
// ---------------------------------------------------------------------------
__global__ void k_wkv_scan(const bhalf* __restrict__ r, const bhalf* __restrict__ k,
    const bhalf* __restrict__ v, const float* __restrict__ df,
    const float* __restrict__ u, const float* __restrict__ S0,
    float* __restrict__ Sout, bhalf* __restrict__ o){
  int bh = blockIdx.x;
  int b = bh >> 5, h = bh & 31;
  int vc = threadIdx.x;                 // 0..63
  __shared__ float rs[64], ks[64], ws[64], us[64];
  us[vc] = u[h*64 + vc];
  float S[64];
  const float* s0 = S0 + (size_t)bh*64*64;
  #pragma unroll
  for (int kk = 0; kk < 64; kk++) S[kk] = s0[kk*64 + vc];
  size_t base = ((size_t)b*TT)*CC + h*64 + vc;
  for (int t = 0; t < TT; t++){
    __syncthreads();
    rs[vc] = b2f(r[base]); ks[vc] = b2f(k[base]); ws[vc] = df[base];
    float vt = b2f(v[base]);
    __syncthreads();
    float acc = 0.f;
    #pragma unroll
    for (int kk = 0; kk < 64; kk++){
      float kv = ks[kk]*vt;
      acc = fmaf(rs[kk], S[kk] + us[kk]*kv, acc);
      S[kk] = fmaf(ws[kk], S[kk], kv);
    }
    o[base] = f2b(acc);
    base += CC;
  }
  float* so = Sout + (size_t)bh*64*64;
  #pragma unroll
  for (int kk = 0; kk < 64; kk++) so[kk*64 + vc] = S[kk];
}

// ---------------------------------------------------------------------------
// Per-head GroupNorm * silu(g) -> bf16 (A-matrix for Wo GEMM). 1 wave / (m,h).
// ---------------------------------------------------------------------------
__global__ void k_gnorm_mul(const bhalf* __restrict__ o, const bhalf* __restrict__ g,
    const float* __restrict__ gw, const float* __restrict__ gbv,
    bhalf* __restrict__ out){
  int mh = blockIdx.x;
  size_t m = (size_t)(mh >> 5);
  int h = mh & 31;
  size_t base = m*CC + h*64;
  int l = threadIdx.x;                  // 0..31, each lane handles n and n+32
  float a0 = b2f(o[base + l]), a1 = b2f(o[base + 32 + l]);
  float s = a0 + a1, ss = a0*a0 + a1*a1;
  #pragma unroll
  for (int off = 16; off > 0; off >>= 1){
    s  += __shfl_xor(s,  off, 32);
    ss += __shfl_xor(ss, off, 32);
  }
  float mu  = s * (1.f/64.f);
  float inv = rsqrtf(ss*(1.f/64.f) - mu*mu + GN_EPS);
  float n0 = (a0 - mu)*inv*gw[h*64 + l]      + gbv[h*64 + l];
  float n1 = (a1 - mu)*inv*gw[h*64 + 32 + l] + gbv[h*64 + 32 + l];
  out[base + l]      = f2b(n0 * b2f(g[base + l]));
  out[base + 32 + l] = f2b(n1 * b2f(g[base + 32 + l]));
}

__global__ void k_lx(const float* __restrict__ x, float* __restrict__ lx){
  int i = blockIdx.x*256 + threadIdx.x;       // < BB*CC
  int b = i >> 11, c = i & (CC-1);
  lx[i] = x[((size_t)b*TT + (TT-1))*CC + c];
}

// ---------------------------------------------------------------------------
extern "C" void kernel_launch(void* const* d_in, const int* in_sizes, int n_in,
                              void* d_out, int out_size, void* d_ws, size_t ws_size,
                              hipStream_t stream){
  const float* x      = (const float*)d_in[0];
  const float* shift  = (const float*)d_in[1];
  const float* wkv0   = (const float*)d_in[2];
  const float* maa_x  = (const float*)d_in[3];
  const float* maa_w  = (const float*)d_in[4];
  const float* maa_k  = (const float*)d_in[5];
  const float* maa_v  = (const float*)d_in[6];
  const float* maa_r  = (const float*)d_in[7];
  const float* maa_g  = (const float*)d_in[8];
  const float* maa_w1 = (const float*)d_in[9];   // [C,160]
  const float* maa_w2 = (const float*)d_in[10];  // [5,32,C]
  const float* tdecay = (const float*)d_in[11];  // [C]
  const float* dw1    = (const float*)d_in[12];  // [C,64]
  const float* dw2    = (const float*)d_in[13];  // [64,C]
  const float* faaaa  = (const float*)d_in[14];  // [H,N]
  const float* Wr     = (const float*)d_in[15];
  const float* Wk     = (const float*)d_in[16];
  const float* Wv     = (const float*)d_in[17];
  const float* Wg     = (const float*)d_in[18];
  const float* Wo     = (const float*)d_in[19];
  const float* gnw    = (const float*)d_in[20];
  const float* gnb    = (const float*)d_in[21];

  float* outY  = (float*)d_out;                  // [B,T,C]
  float* outLx = outY + (size_t)BB*TT*CC;        // [B,C]
  float* outS  = outLx + (size_t)BB*CC;          // [B,H,N,N]

  char* wsp = (char*)d_ws;
  auto take = [&](size_t bytes)->char*{
    char* p = wsp; wsp += (bytes + 255) & ~(size_t)255; return p;
  };
  bhalf* WtR   = (bhalf*)take((size_t)CC*CC*2);
  bhalf* WtK   = (bhalf*)take((size_t)CC*CC*2);
  bhalf* WtV   = (bhalf*)take((size_t)CC*CC*2);
  bhalf* WtG   = (bhalf*)take((size_t)CC*CC*2);
  bhalf* WtO   = (bhalf*)take((size_t)CC*CC*2);
  bhalf* WtM1  = (bhalf*)take((size_t)160*CC*2);     // [160][2048]
  bhalf* WtD1  = (bhalf*)take((size_t)DDECC*CC*2);   // [64][2048]
  bhalf* WtD2  = (bhalf*)take((size_t)CC*DDECC*2);   // [2048][64]
  bhalf* xxx   = (bhalf*)take((size_t)MM*CC*2);
  bhalf* t5    = (bhalf*)take((size_t)MM*160*2);
  bhalf* xw    = (bhalf*)take((size_t)MM*CC*2);
  bhalf* xk    = (bhalf*)take((size_t)MM*CC*2);
  bhalf* xv    = (bhalf*)take((size_t)MM*CC*2);
  bhalf* xr    = (bhalf*)take((size_t)MM*CC*2);
  bhalf* xg    = (bhalf*)take((size_t)MM*CC*2);
  bhalf* tdec  = (bhalf*)take((size_t)MM*DDECC*2);
  float* dfp   = (float*)take((size_t)MM*CC*4);
  bhalf* rb    = (bhalf*)take((size_t)MM*CC*2);
  bhalf* kb    = (bhalf*)take((size_t)MM*CC*2);
  bhalf* vb    = (bhalf*)take((size_t)MM*CC*2);
  bhalf* gbuf  = (bhalf*)take((size_t)MM*CC*2);
  bhalf* ob    = (bhalf*)take((size_t)MM*CC*2);
  bhalf* gnx   = (bhalf*)take((size_t)MM*CC*2);

  dim3 blk(256);

  // Weight convert+transpose to bf16 Bt[N][K]
  k_transpose_bf16<<<dim3(CC/32,    CC/32),    blk, 0, stream>>>(Wr,     WtR,  CC,    CC);
  k_transpose_bf16<<<dim3(CC/32,    CC/32),    blk, 0, stream>>>(Wk,     WtK,  CC,    CC);
  k_transpose_bf16<<<dim3(CC/32,    CC/32),    blk, 0, stream>>>(Wv,     WtV,  CC,    CC);
  k_transpose_bf16<<<dim3(CC/32,    CC/32),    blk, 0, stream>>>(Wg,     WtG,  CC,    CC);
  k_transpose_bf16<<<dim3(CC/32,    CC/32),    blk, 0, stream>>>(Wo,     WtO,  CC,    CC);
  k_transpose_bf16<<<dim3(160/32,   CC/32),    blk, 0, stream>>>(maa_w1, WtM1, CC,    160);
  k_transpose_bf16<<<dim3(DDECC/32, CC/32),    blk, 0, stream>>>(dw1,    WtD1, CC,    DDECC);
  k_transpose_bf16<<<dim3(CC/32,    DDECC/32), blk, 0, stream>>>(dw2,    WtD2, DDECC, CC);

  // Token shift -> xxx (bf16)
  k_shift_mix<<<(MM*CC)/256, blk, 0, stream>>>(x, shift, maa_x, xxx);

  // t5 = tanh(xxx @ maa_w1)   [M,160], bf16   (narrow N -> MT=1, NT=2)
  k_gemm<1,2,1><<<dim3(160/32, MM/128), blk, 0, stream>>>(xxx, WtM1, MM, 160, CC,
                                                          nullptr, t5, nullptr);

  // 5 mixes -> xw,xk,xv,xr,xg (bf16 A-matrices)
  k_mix_combine<<<MM*(CC/256), blk, 0, stream>>>(x, shift, t5, maa_w2,
      maa_w, maa_k, maa_v, maa_r, maa_g, xw, xk, xv, xr, xg);

  // Decay path: tdec = tanh(xw @ dw1); df = exp(-exp(time_decay + tdec @ dw2))
  k_gemm<2,4,1><<<dim3(DDECC/64, MM/256), blk, 0, stream>>>(xw,   WtD1, MM, DDECC, CC,
                                                            nullptr, tdec, nullptr);
  k_gemm<2,4,3><<<dim3(CC/64,    MM/256), blk, 0, stream>>>(tdec, WtD2, MM, CC, DDECC,
                                                            dfp, nullptr, tdecay);

  // Projections r,k,v (bf16) and g = silu(xg@Wg) (bf16); 32x64 wave tiles
  k_gemm<2,4,4><<<dim3(CC/64, MM/256), blk, 0, stream>>>(xr, WtR, MM, CC, CC, nullptr, rb,   nullptr);
  k_gemm<2,4,4><<<dim3(CC/64, MM/256), blk, 0, stream>>>(xk, WtK, MM, CC, CC, nullptr, kb,   nullptr);
  k_gemm<2,4,4><<<dim3(CC/64, MM/256), blk, 0, stream>>>(xv, WtV, MM, CC, CC, nullptr, vb,   nullptr);
  k_gemm<2,4,2><<<dim3(CC/64, MM/256), blk, 0, stream>>>(xg, WtG, MM, CC, CC, nullptr, gbuf, nullptr);

  // WKV recurrence (writes o and final state directly into d_out state region)
  k_wkv_scan<<<BB*HH, 64, 0, stream>>>(rb, kb, vb, dfp, faaaa, wkv0, outS, ob);

  // GroupNorm per head, * g
  k_gnorm_mul<<<MM*HH, 32, 0, stream>>>(ob, gbuf, gnw, gnb, gnx);

  // out = gnx @ Wo  (f32 into d_out)
  k_gemm<2,4,0><<<dim3(CC/64, MM/256), blk, 0, stream>>>(gnx, WtO, MM, CC, CC, outY, nullptr, nullptr);

  // lx = x[:, -1, :]
  k_lx<<<(BB*CC)/256, blk, 0, stream>>>(x, outLx);
}
